// IndexedBeamConvPCC_82566451298960
// MI455X (gfx1250) — matrix operand order repaired and checked by reading, not compiled, not run on hardware
//
#include <hip/hip_runtime.h>
#include <hip/hip_bf16.h>
#include <cstdint>

typedef __attribute__((ext_vector_type(16))) _Float16 v16h;
typedef __attribute__((ext_vector_type(8)))  float    v8f;

#define NPTS   100000
#define PP     12
#define CIN    11
#define KK     64
#define WIN    3
#define OO     4
#define HEADN  2
#define BINS   2
#define NPAD   131072          // next pow2 >= NPTS
#define IDMASK 0x1FFFFu        // 17 bits of point id
#define LNEPS  1e-3f
#define SORT_BLK 2048          // elements per fused-sort block (16KB LDS)

// ---------------------------------------------------------------- utilities
__global__ void zero_f32v4_kernel(float4* __restrict__ p, long n4) {
    long t = (long)blockIdx.x * blockDim.x + threadIdx.x;
    if (t < n4) p[t] = make_float4(0.f, 0.f, 0.f, 0.f);
}

// ---------------------------------------------------------------- sorting
// Packed key: (morton << 17) | point_id  -> stable argsort via id tiebreak.
__global__ void keys_kernel(const long long* __restrict__ pos,
                            unsigned long long* __restrict__ buf) {
    int t = blockIdx.x * blockDim.x + threadIdx.x;  // covers 3*NPAD
    if (t >= 3 * NPAD) return;
    int o = t / NPAD;        // roll shift i = o (produces inv[o+1])
    int j = t - o * NPAD;
    unsigned long long v = ~0ULL;
    if (j < NPTS) {
        // rolled[d] = pos[(d - o) mod 3]; OFFSET=[1,3,3]; SHIFTS=[0,12,24]
        long long r0 = pos[j * 3 + ((0 - o) + 3) % 3];
        long long r1 = pos[j * 3 + ((1 - o) + 3) % 3];
        long long r2 = pos[j * 3 + ((2 - o) + 3) % 3];
        unsigned long long key = (unsigned long long)r0
                               | ((unsigned long long)(r1 / 3) << 12)
                               | ((unsigned long long)(r2 / 3) << 24);
        v = (key << 17) | (unsigned int)j;
    }
    buf[t] = v;
}

// Global phase: used only while j > SORT_BLK/2.
__global__ void bitonic_global_kernel(unsigned long long* __restrict__ buf, int kk, int jj) {
    int t = blockIdx.x * blockDim.x + threadIdx.x;
    if (t >= 3 * NPAD) return;
    int o = t / NPAD;
    int i = t - o * NPAD;
    int ixj = i ^ jj;
    if (ixj <= i) return;
    unsigned long long* b = buf + (size_t)o * NPAD;
    unsigned long long a = b[i], c = b[ixj];
    bool up = ((i & kk) == 0);
    if (up ? (a > c) : (a < c)) { b[i] = c; b[ixj] = a; }
}

// Fused LDS phase: finishes all j-steps from jstart (<= SORT_BLK/2) down to 1
// inside one 2048-element block (partners stay within the aligned block).
__global__ __launch_bounds__(SORT_BLK / 2)
void bitonic_fused_kernel(unsigned long long* __restrict__ buf, int kk, int jstart) {
    __shared__ unsigned long long s[SORT_BLK];
    const int bpr = NPAD / SORT_BLK;               // blocks per region
    int o   = blockIdx.x / bpr;
    int blk = blockIdx.x - o * bpr;
    unsigned long long* b = buf + (size_t)o * NPAD + (size_t)blk * SORT_BLK;
    int t = threadIdx.x;                           // 0..1023
    s[t] = b[t];
    s[t + SORT_BLK / 2] = b[t + SORT_BLK / 2];
    __syncthreads();
    const int gbase = blk * SORT_BLK;              // region-local base index
    for (int j = jstart; j > 0; j >>= 1) {
        int i = ((t & ~(j - 1)) << 1) | (t & (j - 1));
        bool up = (((gbase + i) & kk) == 0);
        unsigned long long x = s[i], y = s[i + j];
        if (up ? (x > y) : (x < y)) { s[i] = y; s[i + j] = x; }
        __syncthreads();
    }
    b[t] = s[t];
    b[t + SORT_BLK / 2] = s[t + SORT_BLK / 2];
}

__global__ void extract_index_kernel(const unsigned long long* __restrict__ buf,
                                     int* __restrict__ index) {
    int t = blockIdx.x * blockDim.x + threadIdx.x;   // covers 4*NPTS
    if (t >= OO * NPTS) return;
    int o = t / NPTS;
    int j = t - o * NPTS;
    if (o == 0) index[t] = j;                        // identity ordering
    else        index[t] = (int)(buf[(size_t)(o - 1) * NPAD + j] & IDMASK);
}

// ---------------------------------------------------------------- weight frags
// Layout: [o][w][cc][kt][lane][e] halves; B-fragment element for lane L, half e:
//   K(=input channel within 32-chunk) = (L/16)*16 + e ; N(=output channel) = L%16
__global__ void prep_wfrag_kernel(const float* __restrict__ W,  // [O,WIN,C,K]
                                  _Float16* __restrict__ frag, int C) {
    int CC = (C + 31) / 32;
    int total = OO * WIN * CC * (KK / 16) * 32 * 16;
    int t = blockIdx.x * blockDim.x + threadIdx.x;
    if (t >= total) return;
    int e    = t & 15;
    int lane = (t >> 4) & 31;
    int rest = t >> 9;
    int kt = rest % (KK / 16); rest /= (KK / 16);
    int cc = rest % CC;        rest /= CC;
    int w  = rest % WIN;
    int o  = rest / WIN;
    int c = cc * 32 + (lane >> 4) * 16 + e;
    int k = kt * 16 + (lane & 15);
    float v = (c < C) ? W[(((o * WIN + w) * C + c) * KK) + k] : 0.f;
    frag[t] = (_Float16)v;
}

// ---------------------------------------------------------------- WMMA conv
// One wave per (16-point block, p, ordering). y[n,p,k] = sum_w,c Xs[n+w-1,p,c]*W[o,w,c,k]
// A-fragment built once per (w,cc) and reused across all 4 K-tiles.
template <typename T, int C>
__global__ __launch_bounds__(32)
void conv_wmma_kernel(const T* __restrict__ Xcur,        // [N,P,C]
                      const _Float16* __restrict__ wfrag,
                      const int* __restrict__ index,     // [O,N]
                      float* __restrict__ acc) {         // [N,P,K]
    constexpr int CC = (C + 31) / 32;
    constexpr int KT = KK / 16;
    __shared__ _Float16 sA[18][CC * 32];

    const int lane = threadIdx.x;
    const int base = blockIdx.x * 16;
    const int p    = blockIdx.y;
    const int o    = blockIdx.z;
    const int* idx = index + o * NPTS;

    // Pull the next tile's permutation entries toward the WGP ahead of use.
    __builtin_prefetch(idx + base + 17, 0, 0);

    // Stage 18 gathered rows (base-1 .. base+16) x C channels -> f16, zero-pad.
    for (int t = lane; t < 18 * CC * 32; t += 32) {
        int r = t / (CC * 32);
        int c = t - r * (CC * 32);
        int srow = base - 1 + r;
        float v = 0.f;
        if (srow >= 0 && srow < NPTS && c < C) {
            int id = idx[srow];
            v = (float)Xcur[((long)id * PP + p) * C + c];
        }
        sA[r][c] = (_Float16)v;
    }
    __syncthreads();

    const int hi = lane >> 4;   // lane group
    const int m  = lane & 15;   // M (points) / N (output-k) sub-index

    v8f d[KT];
    #pragma unroll
    for (int kt = 0; kt < KT; ++kt) d[kt] = (v8f){};

    #pragma unroll
    for (int w = 0; w < WIN; ++w) {
        #pragma unroll
        for (int cc = 0; cc < CC; ++cc) {
            // A fragment: 16x32 f16, layout: M=lane%16, K=(lane/16)*16+e
            v16h a;
            #pragma unroll
            for (int e = 0; e < 16; ++e) {
                int kkc = (hi << 4) + e;
                a[e] = sA[m + w][cc * 32 + kkc];
            }
            #pragma unroll
            for (int kt = 0; kt < KT; ++kt) {
                v16h b = *(const v16h*)(wfrag +
                    (((((o * WIN + w) * CC + cc) * KT + kt) * 32 + lane) << 4));
                d[kt] = __builtin_amdgcn_wmma_f32_16x16x32_f16(
                            false, a, false, b, (short)0, d[kt], false, false);
            }
        }
    }

    // Scatter-add per C/D layout: VGPR r -> M = r + 8*hi, N = lane%16
    #pragma unroll
    for (int kt = 0; kt < KT; ++kt) {
        #pragma unroll
        for (int r = 0; r < 8; ++r) {
            int srow = base + r + (hi << 3);
            if (srow < NPTS) {
                int id = idx[srow];
                atomicAdd(&acc[((long)id * PP + p) * KK + kt * 16 + m], d[kt][r]);
            }
        }
    }
}

// ---------------------------------------------------------------- bias+relu+LN
// Writes f16 activations: halves HBM gather traffic for the next layer's
// 4-ordering gathers (WMMA consumes f16 anyway; LN math stays in f32).
__global__ void post_kernel(const float* __restrict__ acc,
                            const float* __restrict__ bias,
                            const float* __restrict__ g,
                            const float* __restrict__ bb,
                            _Float16* __restrict__ Xnext) {
    long t = (long)blockIdx.x * blockDim.x + threadIdx.x;  // one per (n,p)
    if (t >= (long)NPTS * PP) return;
    const float* row = acc + t * KK;
    float y[KK];
    float mean = 0.f;
    #pragma unroll 8
    for (int k = 0; k < KK; ++k) {
        float v = row[k] + bias[k];
        v = v > 0.f ? v : 0.f;
        y[k] = v;
        mean += v;
    }
    mean *= (1.f / KK);
    float var = 0.f;
    #pragma unroll 8
    for (int k = 0; k < KK; ++k) { float dlt = y[k] - mean; var += dlt * dlt; }
    var *= (1.f / KK);
    float inv = rsqrtf(var + LNEPS);
    _Float16* out = Xnext + t * KK;
    #pragma unroll 8
    for (int k = 0; k < KK; ++k)
        out[k] = (_Float16)((y[k] - mean) * inv * g[k] + bb[k]);
}

// ---------------------------------------------------------------- distill
__device__ __forceinline__ float softplus_f(float v) {
    float a = fabsf(v);
    return log1pf(__expf(-a)) + fmaxf(v, 0.f);
}

template <typename T, int C>
__global__ void distill_kernel(const T* __restrict__ Xin,    // [N,P,C]
                               const int* __restrict__ target,
                               const float* __restrict__ W1, // [C,2]
                               const float* __restrict__ b1, // [2]
                               const float* __restrict__ W2, // [P*2,2]
                               const float* __restrict__ b2, // [2]
                               float* __restrict__ Y) {      // [N,2] accumulated
    int t = blockIdx.x * blockDim.x + threadIdx.x;
    if (t >= NPTS) return;
    int id = target[t];
    const T* x = Xin + (long)id * PP * C;
    float l0 = b2[0], l1 = b2[1];
    for (int p = 0; p < PP; ++p) {
        #pragma unroll
        for (int j = 0; j < HEADN; ++j) {
            float s = b1[j];
            for (int c = 0; c < C; ++c) s += (float)x[p * C + c] * W1[c * HEADN + j];
            float h = softplus_f(s);
            l0 += h * W2[(p * HEADN + j) * BINS + 0];
            l1 += h * W2[(p * HEADN + j) * BINS + 1];
        }
    }
    Y[t * BINS + 0] += l0;
    Y[t * BINS + 1] += l1;
}

__global__ void softmax_kernel(const float* __restrict__ Y, float* __restrict__ out) {
    int t = blockIdx.x * blockDim.x + threadIdx.x;
    if (t >= NPTS) return;
    float a = Y[t * BINS + 0], b = Y[t * BINS + 1];
    float mx = fmaxf(a, b);
    float ea = __expf(a - mx), eb = __expf(b - mx);
    float inv = 1.f / (ea + eb);
    out[t * BINS + 0] = ea * inv;
    out[t * BINS + 1] = eb * inv;
}

// ---------------------------------------------------------------- launch
static inline size_t align_up(size_t x) { return (x + 255) & ~(size_t)255; }

extern "C" void kernel_launch(void* const* d_in, const int* in_sizes, int n_in,
                              void* d_out, int out_size, void* d_ws, size_t ws_size,
                              hipStream_t stream) {
    const float*     X0     = (const float*)d_in[0];
    const long long* pos    = (const long long*)d_in[1];
    const int*       target = (const int*)d_in[2];
    const float* convW[3] = {(const float*)d_in[3],  (const float*)d_in[4],  (const float*)d_in[5]};
    const float* convB[3] = {(const float*)d_in[6],  (const float*)d_in[7],  (const float*)d_in[8]};
    const float* ng[3]    = {(const float*)d_in[9],  (const float*)d_in[10], (const float*)d_in[11]};
    const float* nb[3]    = {(const float*)d_in[12], (const float*)d_in[13], (const float*)d_in[14]};
    const float* dW1[4]   = {(const float*)d_in[15], (const float*)d_in[16], (const float*)d_in[17], (const float*)d_in[18]};
    const float* db1[4]   = {(const float*)d_in[19], (const float*)d_in[20], (const float*)d_in[21], (const float*)d_in[22]};
    const float* dW2[4]   = {(const float*)d_in[23], (const float*)d_in[24], (const float*)d_in[25], (const float*)d_in[26]};
    const float* db2[4]   = {(const float*)d_in[27], (const float*)d_in[28], (const float*)d_in[29], (const float*)d_in[30]};

    // Workspace carve-up
    size_t off = 0;
    size_t oY     = off; off = align_up(off + (size_t)NPTS * BINS * 4);
    size_t oIdx   = off; off = align_up(off + (size_t)OO * NPTS * 4);
    size_t oSort  = off; off = align_up(off + (size_t)3 * NPAD * 8);
    size_t oFrag  = off; off = align_up(off + (size_t)OO * WIN * 2 * (KK / 16) * 32 * 16 * 2);
    size_t oXbuf  = off; off = align_up(off + (size_t)NPTS * PP * KK * 2);   // f16 activations
    size_t oAcc   = off; off = align_up(off + (size_t)NPTS * PP * KK * 4);
    if (ws_size < off) return;   // not enough scratch; no-op

    char* ws = (char*)d_ws;
    float*              Y     = (float*)(ws + oY);
    int*                index = (int*)(ws + oIdx);
    unsigned long long* sbuf  = (unsigned long long*)(ws + oSort);
    _Float16*           frag  = (_Float16*)(ws + oFrag);
    _Float16*           Xbuf  = (_Float16*)(ws + oXbuf);
    float*              accb  = (float*)(ws + oAcc);
    float*              out   = (float*)d_out;

    const int TB = 256;
    long nY4   = (long)NPTS * BINS / 4;
    long nAcc4 = (long)NPTS * PP * KK / 4;

    // 1) zero Y; build keys; bitonic sort (stable via id tiebreak); extract index
    zero_f32v4_kernel<<<(int)((nY4 + TB - 1) / TB), TB, 0, stream>>>((float4*)Y, nY4);
    int nSortT = 3 * NPAD;
    keys_kernel<<<(nSortT + TB - 1) / TB, TB, 0, stream>>>(pos, sbuf);
    for (int k = 2; k <= NPAD; k <<= 1) {
        int j = k >> 1;
        for (; j > SORT_BLK / 2; j >>= 1)
            bitonic_global_kernel<<<(nSortT + TB - 1) / TB, TB, 0, stream>>>(sbuf, k, j);
        bitonic_fused_kernel<<<3 * (NPAD / SORT_BLK), SORT_BLK / 2, 0, stream>>>(sbuf, k, j);
    }
    extract_index_kernel<<<(OO * NPTS + TB - 1) / TB, TB, 0, stream>>>(sbuf, index);

    // 2) distill 0 on raw input
    distill_kernel<float, CIN><<<(NPTS + TB - 1) / TB, TB, 0, stream>>>(
        X0, target, dW1[0], db1[0], dW2[0], db2[0], Y);

    dim3 cgrid((NPTS + 15) / 16, PP, OO);
    int fragT = OO * WIN * 2 * (KK / 16) * 32 * 16;

    // 3) layer 0 (f32 input, C = 11)
    prep_wfrag_kernel<<<(fragT + TB - 1) / TB, TB, 0, stream>>>(convW[0], frag, CIN);
    zero_f32v4_kernel<<<(int)((nAcc4 + TB - 1) / TB), TB, 0, stream>>>((float4*)accb, nAcc4);
    conv_wmma_kernel<float, CIN><<<cgrid, 32, 0, stream>>>(X0, frag, index, accb);
    post_kernel<<<(int)(((long)NPTS * PP + TB - 1) / TB), TB, 0, stream>>>(
        accb, convB[0], ng[0], nb[0], Xbuf);
    distill_kernel<_Float16, KK><<<(NPTS + TB - 1) / TB, TB, 0, stream>>>(
        Xbuf, target, dW1[1], db1[1], dW2[1], db2[1], Y);

    // 4) layers 1..2 (f16 activations, C = 64), ping buffer reused in place
    for (int L = 1; L < 3; ++L) {
        prep_wfrag_kernel<<<(fragT + TB - 1) / TB, TB, 0, stream>>>(convW[L], frag, KK);
        zero_f32v4_kernel<<<(int)((nAcc4 + TB - 1) / TB), TB, 0, stream>>>((float4*)accb, nAcc4);
        conv_wmma_kernel<_Float16, KK><<<cgrid, 32, 0, stream>>>(Xbuf, frag, index, accb);
        post_kernel<<<(int)(((long)NPTS * PP + TB - 1) / TB), TB, 0, stream>>>(
            accb, convB[L], ng[L], nb[L], Xbuf);
        distill_kernel<_Float16, KK><<<(NPTS + TB - 1) / TB, TB, 0, stream>>>(
            Xbuf, target, dW1[L + 1], db1[L + 1], dW2[L + 1], db2[L + 1], Y);
    }

    // 5) softmax
    softmax_kernel<<<(NPTS + TB - 1) / TB, TB, 0, stream>>>(Y, out);
}
